// DynaConv_76708115906720
// MI455X (gfx1250) — compile-verified, matrix operand order checked by reading, not run on hardware
//
#include <hip/hip_runtime.h>
#include <math.h>

typedef __attribute__((ext_vector_type(16))) _Float16 v16h;
typedef __attribute__((ext_vector_type(8)))  float    v8f;

#define CIN   32
#define COUT  32
#define HID   32
#define CKK   288
#define KC    9          // 288 / 32 k-chunks per WMMA K
#define HH    64
#define WW    64
#define NPIX_TILE 64     // pixels per workgroup (4 WMMA col tiles)
#define NPT   4          // pixel sub-tiles of 16
#define NTHREADS  256    // 8 wave32

// fragment-major packed weight sizes (halves)
#define NW2F   (64 * KC * 512)   // 294912 : W2r rows 1024 -> 64 row-tiles
#define NSMALL (2  * KC * 512)   // 9216   : 32 rows -> 2 row-tiles

__device__ __forceinline__ v8f wmma_f16(v16h a, v16h b, v8f c) {
  return __builtin_amdgcn_wmma_f32_16x16x32_f16(false, a, false, b,
                                                (short)0, c, false, false);
}

// lane/i -> K index inside a 32-wide chunk, per the 16-bit A-matrix layout:
// lanes 0-15 hold K {0-7,16-23}; lanes 16-31 hold K {8-15,24-31}.
__device__ __forceinline__ int frag_klocal(int lane, int i) {
  return (lane < 16) ? ((i < 8) ? i : i + 8)
                     : ((i < 8) ? i + 8 : i + 16);
}

// ---- one-time repack of W1 / W2 / b2 into f16 fragment-major layout ----
__global__ void dynaconv_pack(const float* __restrict__ W1,
                              const float* __restrict__ W2,
                              const float* __restrict__ b2,
                              _Float16* __restrict__ w2f,
                              _Float16* __restrict__ w1f,
                              _Float16* __restrict__ b2f) {
  int gid = blockIdx.x * blockDim.x + threadIdx.x;
  if (gid >= NW2F + 2 * NSMALL) return;
  int i    = gid & 15;
  int lane = (gid >> 4) & 31;          // section offsets are multiples of 512
  int k_local = frag_klocal(lane, i);
  int m = lane & 15;
  if (gid < NW2F) {
    int blk = gid >> 9;                // t*KC + kc
    int t = blk / KC, kc = blk % KC;
    int r = t * 16 + m;                // W2r row = o*32 + h (h minor)
    int o = r >> 5, h = r & 31;
    int j = kc * 32 + k_local;
    w2f[gid] = (_Float16)W2[(o * CKK + j) * HID + h];
  } else if (gid < NW2F + NSMALL) {
    int g = gid - NW2F;
    int blk = g >> 9;
    int t = blk / KC, kc = blk % KC;
    int hrow = t * 16 + m;
    int j = kc * 32 + k_local;
    w1f[g] = (_Float16)W1[hrow * CKK + j];
  } else {
    int g = gid - NW2F - NSMALL;
    int blk = g >> 9;
    int t = blk / KC, kc = blk % KC;
    int orow = t * 16 + m;
    int j = kc * 32 + k_local;
    b2f[g] = (_Float16)b2[orow * CKK + j];
  }
}

// ---- fused dynamic-conv kernel: one WG per 64-pixel tile ----
__global__ void __launch_bounds__(NTHREADS)
dynaconv_main(const float* __restrict__ x,
              const float* __restrict__ b1,
              const float* __restrict__ bias,
              const _Float16* __restrict__ w2f,
              const _Float16* __restrict__ w1f,
              const _Float16* __restrict__ b2f,
              float* __restrict__ out) {
  __shared__ __attribute__((aligned(32))) _Float16 ufrag[KC * NPT * 32 * 16]; // 36KB, B-frag layout
  __shared__ float Hl[HID  * NPIX_TILE];   // tanh hidden tile, 8KB
  __shared__ float Cl[COUT * NPIX_TILE];   // b2-path + bias tile, 8KB

  const int tid  = threadIdx.x;
  const int wg   = blockIdx.x;             // 128 tiles of 64 pixels
  const int lane = tid & 31;
  const int w    = tid >> 5;               // wave id 0..7

  // ---- build unfold tile U[288 x 64] directly in WMMA B-fragment layout ----
  for (int idx = tid; idx < CKK * NPIX_TILE; idx += NTHREADS) {
    int j = idx >> 6;                      // 0..287  (c*9 + ki*3 + kj)
    int p = idx & 63;                      // pixel within tile
    int P = wg * NPIX_TILE + p;
    int b = P >> 12;
    int l = P & 4095;
    int y = l >> 6, xx = l & 63;
    int c = j / 9, rem = j % 9;
    int ki = rem / 3, kj = rem % 3;
    int iy = y + ki - 1, ix = xx + kj - 1;
    float val = 0.f;
    if ((unsigned)iy < HH && (unsigned)ix < WW)
      val = x[((b * CIN + c) * HH + iy) * WW + ix];
    int kc = j >> 5, k_local = j & 31;
    int pt = p >> 4, n = p & 15;
    int lnf, ii;                           // inverse of frag_klocal
    if (k_local < 8)       { lnf = n;      ii = k_local;      }
    else if (k_local < 16) { lnf = n + 16; ii = k_local - 8;  }
    else if (k_local < 24) { lnf = n;      ii = k_local - 8;  }
    else                   { lnf = n + 16; ii = k_local - 16; }
    ufrag[((kc * NPT + pt) * 32 + lnf) * 16 + ii] = (_Float16)val;
  }
  __syncthreads();

  const int n    = lane & 15;
  const int mofs = (lane < 16) ? 0 : 8;

  // ---- small GEMMs: H = tanh(W1@U + b1), C = b2r@U + bias (2 tiles/wave) ----
  {
    int t = w >> 2, pt = w & 3;            // 2 t-tiles x 4 pt-tiles = 8 jobs each
    int p = pt * 16 + n;
    // H job
    {
      v8f acc = {};
#pragma unroll
      for (int kc = 0; kc < KC; ++kc) {
        v16h a  = *(const v16h*)(w1f + ((t * KC + kc) * 32 + lane) * 16);
        v16h bf = *(const v16h*)(&ufrag[((kc * NPT + pt) * 32 + lane) * 16]);
        acc = wmma_f16(a, bf, acc);
      }
#pragma unroll
      for (int v = 0; v < 8; ++v) {
        int row = t * 16 + mofs + v;
        Hl[row * NPIX_TILE + p] = tanhf(acc[v] + b1[row]);
      }
    }
    // C job
    {
      v8f acc = {};
#pragma unroll
      for (int kc = 0; kc < KC; ++kc) {
        v16h a  = *(const v16h*)(b2f + ((t * KC + kc) * 32 + lane) * 16);
        v16h bf = *(const v16h*)(&ufrag[((kc * NPT + pt) * 32 + lane) * 16]);
        acc = wmma_f16(a, bf, acc);
      }
#pragma unroll
      for (int v = 0; v < 8; ++v) {
        int row = t * 16 + mofs + v;
        Cl[row * NPIX_TILE + p] = acc[v] + bias[row];
      }
    }
  }
  __syncthreads();

  // ---- main GEMM G = W2r@U fused with out[o] = sum_h G[(o,h)] * H[h] ----
  // Each wave owns 4 output channels o = w + 8*oo; each A fragment load is
  // reused across all 4 pixel sub-tiles (4 live accumulator sets).
  float partial[4][NPT];
#pragma unroll
  for (int oo = 0; oo < 4; ++oo)
#pragma unroll
    for (int pt = 0; pt < NPT; ++pt) partial[oo][pt] = 0.f;

#pragma unroll
  for (int th = 0; th < 2; ++th) {         // h 0-15 / 16-31 row-tile halves
    // hoist H operands: reused by all 4 output channels of this wave
    float hv[NPT][8];
#pragma unroll
    for (int pt = 0; pt < NPT; ++pt)
#pragma unroll
      for (int v = 0; v < 8; ++v)
        hv[pt][v] = Hl[(th * 16 + mofs + v) * NPIX_TILE + pt * 16 + n];

#pragma unroll
    for (int oo = 0; oo < 4; ++oo) {
      int o = w + oo * 8;
      int t = 2 * o + th;
      v8f acc[NPT] = {};
#pragma unroll
      for (int kc = 0; kc < KC; ++kc) {
        v16h a = *(const v16h*)(w2f + ((t * KC + kc) * 32 + lane) * 16);
#pragma unroll
        for (int pt = 0; pt < NPT; ++pt) {
          v16h bf = *(const v16h*)(&ufrag[((kc * NPT + pt) * 32 + lane) * 16]);
          acc[pt] = wmma_f16(a, bf, acc[pt]);
        }
      }
#pragma unroll
      for (int pt = 0; pt < NPT; ++pt)
#pragma unroll
        for (int v = 0; v < 8; ++v)
          partial[oo][pt] += acc[pt][v] * hv[pt][v];
    }
  }

  // lanes L and L+16 hold rows 0-7 / 8-15 of the same pixel column
#pragma unroll
  for (int oo = 0; oo < 4; ++oo) {
    int o = w + oo * 8;
#pragma unroll
    for (int pt = 0; pt < NPT; ++pt) {
      float total = partial[oo][pt] + __shfl_xor(partial[oo][pt], 16, 32);
      if (lane < 16) {
        int p = pt * 16 + lane;
        int P = wg * NPIX_TILE + p;
        int b = P >> 12, l = P & 4095;
        out[((size_t)(b * COUT + o) << 12) + l] = total + Cl[o * NPIX_TILE + p];
      }
    }
  }
}

extern "C" void kernel_launch(void* const* d_in, const int* in_sizes, int n_in,
                              void* d_out, int out_size, void* d_ws, size_t ws_size,
                              hipStream_t stream) {
  const float* x    = (const float*)d_in[0];
  const float* W1   = (const float*)d_in[1];
  const float* b1   = (const float*)d_in[2];
  const float* W2   = (const float*)d_in[3];
  const float* b2   = (const float*)d_in[4];
  const float* bias = (const float*)d_in[5];
  float* out = (float*)d_out;

  // workspace: packed f16 fragment-major weights (~613 KB total)
  _Float16* w2f = (_Float16*)d_ws;
  _Float16* w1f = w2f + NW2F;
  _Float16* b2f = w1f + NSMALL;

  int npack = NW2F + 2 * NSMALL;           // 313344
  dynaconv_pack<<<(npack + 255) / 256, 256, 0, stream>>>(W1, W2, b2, w2f, w1f, b2f);
  dynaconv_main<<<8192 / NPIX_TILE, NTHREADS, 0, stream>>>(x, b1, bias, w2f, w1f, b2f, out);
}
